// AttentionLayer_37503654429425
// MI455X (gfx1250) — matrix operand order repaired and checked by reading, not compiled, not run on hardware
//
#include <hip/hip_runtime.h>
#include <hip/hip_bf16.h>
#include <math.h>
#include <stdint.h>

// Shapes (fixed by the reference)
#define B_ 32
#define T_ 8192
#define V_ 128
#define A_ 128

typedef __attribute__((ext_vector_type(16))) __bf16 v16bf;
typedef __attribute__((ext_vector_type(8)))  float  v8f;

#define LOG2E_F 1.4426950408889634f

// f32 pair -> packed bf16 dword via one v_perm_b32 (truncating round).
// result bytes: [1:0] = lo[3:2], [3:2] = hi[3:2]
__device__ __forceinline__ uint32_t pack_bf16(float lo, float hi) {
  return __builtin_amdgcn_perm(__float_as_uint(hi), __float_as_uint(lo),
                               0x07060302u);
}

// Branch-free tanh: hardware v_tanh_f32 when the builtin exists, otherwise
// exp2+rcp identity (no exec-divergent range reduction).
__device__ __forceinline__ float fast_tanh(float x) {
#if __has_builtin(__builtin_amdgcn_tanhf)
  return __builtin_amdgcn_tanhf(x);
#else
  const float ax = __builtin_fabsf(x);
  const float e  = __builtin_amdgcn_exp2f(ax * (-2.0f * LOG2E_F)); // e^(-2|x|)
  const float t  = (1.0f - e) * __builtin_amdgcn_rcpf(1.0f + e);
  return __builtin_copysignf(t, x);
#endif
}

// fast exp for x <= 0 (softmax): no overflow possible, plain v_exp_f32
__device__ __forceinline__ float exp_nonpos(float x) {
  return __builtin_amdgcn_exp2f(x * LOG2E_F);
}

// ---------------------------------------------------------------------------
// k0: dec_feat[b,a] = decoder_state[b,:]@W_dec[:,a] + b_dec + b_enc + b_cov
//     (all biases folded once); also zero the context output region.
// grid: 32 blocks x 128 threads
// ---------------------------------------------------------------------------
__global__ void k0_prep(const float* __restrict__ dec, const float* __restrict__ Wdec,
                        const float* __restrict__ b_enc, const float* __restrict__ b_dec,
                        const float* __restrict__ b_cov,
                        float* __restrict__ decfeat, float* __restrict__ ctx_out) {
  const int b = blockIdx.x, a = threadIdx.x;
  float s = 0.f;
  #pragma unroll 8
  for (int k = 0; k < V_; ++k) s = fmaf(dec[b * V_ + k], Wdec[k * A_ + a], s);
  decfeat[b * A_ + a] = s + b_dec[a] + b_enc[a] + b_cov[a];
  ctx_out[b * A_ + a] = 0.f;   // context accumulated with atomics later
}

// ---------------------------------------------------------------------------
// k0b: pre-pack Wk_enc (K=128 x N=128, f32) into WMMA-B bf16 lane layout.
// B tile = 32(K) x 16(N) per (kc, nt). Assumed lane layout (ISA 7.12.2 style):
//   lane l: column N = nt*16 + (l&15); K base = kc*32 + (l<16 ? 0 : 8);
//   dword j (j=0..7) holds packed bf16 pair (K=base+2j+(j>=4?8:0), +1).
// Index: Bp[((kc*8+nt)*32+lane)*8 + j]  -> 8192 dwords total (32 KB).
// ---------------------------------------------------------------------------
__global__ void k0_packB(const float* __restrict__ Wk, uint32_t* __restrict__ Bp) {
  const int idx  = blockIdx.x * blockDim.x + threadIdx.x;  // 0..8191
  const int j    = idx & 7;
  const int lane = (idx >> 3) & 31;
  const int nt   = (idx >> 8) & 7;
  const int kc   = idx >> 11;
  const int n    = nt * 16 + (lane & 15);
  const int kb   = kc * 32 + ((lane < 16) ? 0 : 8) + 2 * j + ((j >= 4) ? 8 : 0);
  Bp[idx] = pack_bf16(Wk[kb * A_ + n], Wk[(kb + 1) * A_ + n]);
}

// ---------------------------------------------------------------------------
// k1: hot kernel.  e[b,t] = sum_a v[a]*tanh(encA[t,a] + decfeat[b,a] + cov[t]*Kc[a])
// Each wave: 16 rows of enc (one 16xK A tile), full N=128 via 8 WMMA N-tiles,
// K=128 via 4 chunks of v_wmma_f32_16x16x32_bf16  -> 32 WMMAs per wave.
// Block = 256 threads = 8 waves = 128 rows;  grid = B*T/128 = 2048 blocks.
// ---------------------------------------------------------------------------
__global__ __launch_bounds__(256) void k1_logits(
    const float* __restrict__ enc, const uint32_t* __restrict__ Bp,
    const float* __restrict__ decfeat, const float* __restrict__ cov,
    const float* __restrict__ Kc, const float* __restrict__ vvec,
    float* __restrict__ e_out) {
  const int lane = threadIdx.x & 31;
  const int wave = threadIdx.x >> 5;
  const int m0   = blockIdx.x * 128 + wave * 16;  // first of this wave's 16 rows
  const int b    = m0 >> 13;                      // /T_ ; tiles never cross b
  const int n_lo = lane & 15;
  const int hi   = lane >> 4;                     // 0 or 1
  const int row  = m0 + n_lo;                     // A-matrix row for this lane
  const int ko   = hi * 8;                        // per-lane K sub-offset (A layout)

  // ---- load + convert A: four 16x32 bf16 chunks (ISA 7.12.2 16-bit A layout)
  union { v16bf v; uint32_t u[8]; } a[4];
  const float* arow = enc + (size_t)row * V_;
  #pragma unroll
  for (int kc = 0; kc < 4; ++kc) {
    const float4 g0a = *(const float4*)(arow + kc * 32 + ko);
    const float4 g0b = *(const float4*)(arow + kc * 32 + ko + 4);
    const float4 g1a = *(const float4*)(arow + kc * 32 + 16 + ko);
    const float4 g1b = *(const float4*)(arow + kc * 32 + 16 + ko + 4);
    a[kc].u[0] = pack_bf16(g0a.x, g0a.y);
    a[kc].u[1] = pack_bf16(g0a.z, g0a.w);
    a[kc].u[2] = pack_bf16(g0b.x, g0b.y);
    a[kc].u[3] = pack_bf16(g0b.z, g0b.w);
    a[kc].u[4] = pack_bf16(g1a.x, g1a.y);
    a[kc].u[5] = pack_bf16(g1a.z, g1a.w);
    a[kc].u[6] = pack_bf16(g1b.x, g1b.y);
    a[kc].u[7] = pack_bf16(g1b.z, g1b.w);
  }

  // coverage for the 8 rows this lane's accumulator slots map to (M = hi*8+d)
  const float4 c0 = *(const float4*)(cov + m0 + hi * 8);
  const float4 c1 = *(const float4*)(cov + m0 + hi * 8 + 4);
  const float covr[8] = {c0.x, c0.y, c0.z, c0.w, c1.x, c1.y, c1.z, c1.w};

  float esum[8] = {0, 0, 0, 0, 0, 0, 0, 0};

  #pragma unroll
  for (int nt = 0; nt < 8; ++nt) {
    v8f acc = {};
    #pragma unroll
    for (int kc = 0; kc < 4; ++kc) {
      union { v16bf v; uint4 q[2]; } bmat;
      const uint4* bp = (const uint4*)(Bp + (((kc * 8 + nt) * 32 + lane) << 3));
      bmat.q[0] = bp[0];
      bmat.q[1] = bp[1];
      acc = __builtin_amdgcn_wmma_f32_16x16x32_bf16(
          /*neg_a=*/false, a[kc].v, /*neg_b=*/false, bmat.v,
          /*c_mod=*/(short)0, acc, /*reuse_a=*/false, /*reuse_b=*/false);
    }
    // epilogue on accumulators: C/D layout -> this lane holds col = nt*16+n_lo,
    // rows M = hi*8 + d in slot d.
    const int   col  = nt * 16 + n_lo;
    const float decc = decfeat[b * A_ + col];
    const float kcc  = Kc[col];
    const float vv   = vvec[col];
    #pragma unroll
    for (int d = 0; d < 8; ++d) {
      const float x = acc[d] + decc + covr[d] * kcc;
      esum[d] += fast_tanh(x) * vv;
    }
  }

  // reduce over the 16 lanes holding the same row (xor 1,2,4,8 — stays in half)
  #pragma unroll
  for (int d = 0; d < 8; ++d) {
    float r = esum[d];
    r += __shfl_xor(r, 1, 32);
    r += __shfl_xor(r, 2, 32);
    r += __shfl_xor(r, 4, 32);
    r += __shfl_xor(r, 8, 32);
    if (n_lo == 0) e_out[m0 + hi * 8 + d] = r;
  }
}

// ---------------------------------------------------------------------------
// k2: per-batch softmax stats: rmax[b] = max_t e ; rsum[b] = sum exp(e-max)*mask
// (mask + renormalization folded into one denominator). grid: 32 x 256
// ---------------------------------------------------------------------------
__global__ __launch_bounds__(256) void k2_stats(const float* __restrict__ e,
                                                const float* __restrict__ mask,
                                                float* __restrict__ rmax,
                                                float* __restrict__ rsum) {
  const int b = blockIdx.x, tid = threadIdx.x;
  __shared__ float red[256];
  float m = -3.0e38f;
  for (int t = tid; t < T_; t += 256) m = fmaxf(m, e[b * T_ + t]);
  red[tid] = m; __syncthreads();
  for (int s = 128; s > 0; s >>= 1) {
    if (tid < s) red[tid] = fmaxf(red[tid], red[tid + s]);
    __syncthreads();
  }
  m = red[0]; __syncthreads();
  float sum = 0.f;
  for (int t = tid; t < T_; t += 256)
    sum += exp_nonpos(e[b * T_ + t] - m) * mask[b * T_ + t];
  red[tid] = sum; __syncthreads();
  for (int s = 128; s > 0; s >>= 1) {
    if (tid < s) red[tid] += red[tid + s];
    __syncthreads();
  }
  if (tid == 0) { rmax[b] = m; rsum[b] = red[0]; }
}

// ---------------------------------------------------------------------------
// k3: attn[b,t] = exp(e-max)*mask/rsum ; coverage_out = coverage + attn
// grid: B*T/256 = 1024 x 256
// ---------------------------------------------------------------------------
__global__ void k3_attn(const float* __restrict__ e, const float* __restrict__ mask,
                        const float* __restrict__ cov_in,
                        const float* __restrict__ rmax, const float* __restrict__ rsum,
                        float* __restrict__ attn, float* __restrict__ cov_out) {
  const int i = blockIdx.x * blockDim.x + threadIdx.x;  // 0..B*T-1
  const int b = i >> 13;
  const float a = exp_nonpos(e[i] - rmax[b]) * mask[i] * (1.f / rsum[b]);
  attn[i] = a;
  cov_out[i] = cov_in[i] + a;
}

// ---------------------------------------------------------------------------
// k4: context[b,v] = sum_t attn[b,t] * enc[b,t,v]  (second streaming pass)
// grid: 32 b * 8 chunks = 256 blocks; block = (ty 0..1) x (vx 0..127)
// ---------------------------------------------------------------------------
__global__ __launch_bounds__(256) void k4_context(const float* __restrict__ enc,
                                                  const float* __restrict__ attn,
                                                  float* __restrict__ ctx) {
  const int b     = blockIdx.x >> 3;
  const int chunk = blockIdx.x & 7;
  const int vx    = threadIdx.x & 127;
  const int ty    = threadIdx.x >> 7;   // 0..1
  const int t0    = chunk * (T_ / 8);
  float acc = 0.f;
  for (int t = t0 + ty; t < t0 + T_ / 8; t += 2)
    acc = fmaf(attn[b * T_ + t], enc[((size_t)b * T_ + t) * V_ + vx], acc);
  atomicAdd(&ctx[b * A_ + vx], acc);
}

// ---------------------------------------------------------------------------
extern "C" void kernel_launch(void* const* d_in, const int* in_sizes, int n_in,
                              void* d_out, int out_size, void* d_ws, size_t ws_size,
                              hipStream_t stream) {
  // inputs (setup_inputs order)
  const float* enc   = (const float*)d_in[0];   // (B,T,V)
  const float* dec   = (const float*)d_in[1];   // (B,V)
  const float* mask  = (const float*)d_in[2];   // (B,T)
  const float* cov   = (const float*)d_in[3];   // (B,T,1,1)
  const float* Wk    = (const float*)d_in[4];   // (V,A)
  const float* b_enc = (const float*)d_in[5];   // (A)
  const float* Wdec  = (const float*)d_in[6];   // (V,A)
  const float* b_dec = (const float*)d_in[7];   // (A)
  const float* Kc    = (const float*)d_in[8];   // (A)
  const float* b_cov = (const float*)d_in[9];   // (A)
  const float* vvec  = (const float*)d_in[10];  // (A)

  // outputs: context (B*V) | attn (B*T) | coverage_out (B*T)
  float* out     = (float*)d_out;
  float* ctx     = out;
  float* attn    = out + B_ * V_;
  float* cov_out = out + B_ * V_ + B_ * T_;

  // workspace layout (floats)
  float* ws      = (float*)d_ws;
  float* e       = ws;                      // B*T       = 262144
  float* rmax    = ws + B_ * T_;            // 32
  float* rsum    = rmax + B_;               // 32
  float* decfeat = rsum + B_;               // B*A = 4096
  uint32_t* Bp   = (uint32_t*)(decfeat + B_ * A_);  // 8192 dwords

  k0_prep   <<<B_, A_, 0, stream>>>(dec, Wdec, b_enc, b_dec, b_cov, decfeat, ctx);
  k0_packB  <<<32, 256, 0, stream>>>(Wk, Bp);
  k1_logits <<<(B_ * T_) / 128, 256, 0, stream>>>(enc, Bp, decfeat, cov, Kc, vvec, e);
  k2_stats  <<<B_, 256, 0, stream>>>(e, mask, rmax, rsum);
  k3_attn   <<<(B_ * T_) / 256, 256, 0, stream>>>(e, mask, cov, rmax, rsum, attn, cov_out);
  k4_context<<<B_ * 8, 256, 0, stream>>>(enc, attn, ctx);
}
